// MambaLayer_3487513445006
// MI455X (gfx1250) — matrix operand order repaired and checked by reading, not compiled, not run on hardware
//
#include <hip/hip_runtime.h>
#include <hip/hip_bf16.h>
#include <math.h>

// ---------------------------------------------------------------------------
// Problem constants (from the reference)
// ---------------------------------------------------------------------------
#define B_SZ     4
#define L_SZ     4096
#define DM_SZ    1024
#define DI_SZ    2048     // D_INNER
#define DS_SZ    16       // D_STATE
#define DC_SZ    4        // D_CONV
#define DTR_SZ   64       // DT_RANK
#define BL_SZ    (B_SZ * L_SZ)          // 16384 rows
#define XP_SZ    (DTR_SZ + 2 * DS_SZ)   // 96

typedef __attribute__((ext_vector_type(16))) __bf16 v16bf;
typedef __attribute__((ext_vector_type(8)))  __bf16 v8bf;
typedef __attribute__((ext_vector_type(8)))  float  v8f;

// ---------------------------------------------------------------------------
// Generic bf16-WMMA GEMM:  C[M,N] = A[M,K] * W[K,N]   (fp32 in memory,
// bf16 in LDS, f32 accumulation via v_wmma_f32_16x16x32_bf16).
// 128x128 block tile, K-step 32, 256 thr = 8 waves, 16x128 strip per wave.
// Software-pipelined: next tile's 8 global_load_b128 are issued into
// registers before the WMMA block; LDS is double-buffered (1 barrier/step).
// epilogue: 0 = none, 1 = softplus(acc + bias[n])
// ---------------------------------------------------------------------------
#define TM 128
#define TN 128
#define TK 32
#define LDS_STRIDE 40   // halfs; 80B/row: 16B-aligned, conflict-free b128

__global__ __launch_bounds__(256)
void gemm_bf16_wmma(const float* __restrict__ A, const float* __restrict__ W,
                    float* __restrict__ C, const float* __restrict__ bias,
                    int M, int N, int K, int lda, int ldw, int ldc,
                    int epilogue)
{
    __shared__ __bf16 As[2][TM][LDS_STRIDE];
    __shared__ __bf16 Bs[2][TN][LDS_STRIDE];

    const int tid  = threadIdx.x;
    const int lane = tid & 31;
    const int wave = tid >> 5;
    const int mBase = blockIdx.y * TM;
    const int nBase = blockIdx.x * TN;

    // per-thread staging coordinates (constant across K-steps)
    const int arow[4] = { (tid + 0*256) >> 3, (tid + 1*256) >> 3,
                          (tid + 2*256) >> 3, (tid + 3*256) >> 3 };
    const int acol    = (tid & 7) << 2;              // same for all 4
    const int bkr [4] = { (tid + 0*256) >> 5, (tid + 1*256) >> 5,
                          (tid + 2*256) >> 5, (tid + 3*256) >> 5 };
    const int bcol    = (tid & 31) << 2;             // same for all 4

    v8f acc[8];
#pragma unroll
    for (int t = 0; t < 8; ++t)
#pragma unroll
        for (int i = 0; i < 8; ++i) acc[t][i] = 0.0f;

    const int lm   = lane & 15;
    const int kbA  = (lane < 16) ? 0 : 8;    // A frag K base (16-bit A layout)
    const int khB  = (lane < 16) ? 0 : 8;    // B frag K base (16-bit B layout)
    const int rowA = wave * 16 + lm;

    float4 aR[4], bR[4];

    // ---- issue loads for K-tile k0 into registers (no waits in between) ----
    auto load_tiles = [&](int k0) {
#pragma unroll
        for (int i = 0; i < 4; ++i)
            aR[i] = *reinterpret_cast<const float4*>(
                &A[(size_t)(mBase + arow[i]) * lda + k0 + acol]);
#pragma unroll
        for (int i = 0; i < 4; ++i) {
            float4 v = make_float4(0.f, 0.f, 0.f, 0.f);
            if (nBase + bcol < N)
                v = *reinterpret_cast<const float4*>(
                    &W[(size_t)(k0 + bkr[i]) * ldw + nBase + bcol]);
            bR[i] = v;
        }
    };

    // ---- convert + store staged registers into LDS buffer p ----------------
    auto store_tiles = [&](int p) {
#pragma unroll
        for (int i = 0; i < 4; ++i) {
            As[p][arow[i]][acol + 0] = (__bf16)aR[i].x;
            As[p][arow[i]][acol + 1] = (__bf16)aR[i].y;
            As[p][arow[i]][acol + 2] = (__bf16)aR[i].z;
            As[p][arow[i]][acol + 3] = (__bf16)aR[i].w;
        }
#pragma unroll
        for (int i = 0; i < 4; ++i) {
            Bs[p][bcol + 0][bkr[i]] = (__bf16)bR[i].x;
            Bs[p][bcol + 1][bkr[i]] = (__bf16)bR[i].y;
            Bs[p][bcol + 2][bkr[i]] = (__bf16)bR[i].z;
            Bs[p][bcol + 3][bkr[i]] = (__bf16)bR[i].w;
        }
    };

    // ---- 8 WMMAs from LDS buffer p -----------------------------------------
    auto compute = [&](int p) {
        union { v16bf v; v8bf h[2]; } afrag;
        afrag.h[0] = *reinterpret_cast<const v8bf*>(&As[p][rowA][kbA]);
        afrag.h[1] = *reinterpret_cast<const v8bf*>(&As[p][rowA][kbA + 16]);
#pragma unroll
        for (int t = 0; t < 8; ++t) {
            int n = t * 16 + lm;
            union { v16bf v; v8bf h[2]; } bfrag;
            bfrag.h[0] = *reinterpret_cast<const v8bf*>(&Bs[p][n][khB]);
            bfrag.h[1] = *reinterpret_cast<const v8bf*>(&Bs[p][n][khB + 16]);
            acc[t] = __builtin_amdgcn_wmma_f32_16x16x32_bf16(
                false, afrag.v, false, bfrag.v, (short)0, acc[t],
                false, false);
        }
    };

    // ---- pipelined main loop -----------------------------------------------
    load_tiles(0);
    store_tiles(0);
    __syncthreads();
    int p = 0;
    for (int k0 = TK; k0 < K; k0 += TK) {
        load_tiles(k0);       // global loads in flight during WMMAs below
        compute(p);
        store_tiles(p ^ 1);   // first register use -> single wait here
        __syncthreads();
        p ^= 1;
    }
    compute(p);

    // ---- store: C/D layout => lanes 0-15 rows 0..7, lanes 16-31 rows 8..15
    const int rBase = mBase + wave * 16 + ((lane < 16) ? 0 : 8);
#pragma unroll
    for (int t = 0; t < 8; ++t) {
        int n = nBase + t * 16 + lm;
        if (n < N) {
            float bval = (epilogue == 1) ? bias[n] : 0.0f;
#pragma unroll
            for (int i = 0; i < 8; ++i) {
                float v = acc[t][i];
                if (epilogue == 1) {
                    v += bval;
                    v = (v > 20.0f) ? v : log1pf(__expf(v));  // softplus
                }
                C[(size_t)(rBase + i) * ldc + n] = v;
            }
        }
    }
}

// ---------------------------------------------------------------------------
// Causal depthwise conv (kernel 4) + bias + SiLU.
// xin lives in xz[:, 0:2048] (row stride 4096).  Out: xc (B,L,DI) contiguous.
// ---------------------------------------------------------------------------
__global__ __launch_bounds__(256)
void conv_silu_kernel(const float* __restrict__ xz,
                      const float* __restrict__ cw,
                      const float* __restrict__ cb,
                      float* __restrict__ xc)
{
    int idx = blockIdx.x * 256 + threadIdx.x;
    if (idx >= B_SZ * L_SZ * DI_SZ) return;
    int d = idx & (DI_SZ - 1);
    int l = (idx / DI_SZ) & (L_SZ - 1);
    int b = idx / (DI_SZ * L_SZ);

    float acc = cb[d];
    const float* wrow = &cw[d * DC_SZ];
#pragma unroll
    for (int j = 0; j < DC_SZ; ++j) {
        int ls = l - (DC_SZ - 1) + j;
        if (ls >= 0)
            acc += wrow[j] * xz[((size_t)(b * L_SZ + ls)) * (2 * DI_SZ) + d];
    }
    xc[idx] = acc / (1.0f + __expf(-acc));   // silu
}

// ---------------------------------------------------------------------------
// Fused selective scan + D-skip + gating by silu(z).
// One lane per (b,d); h[16] lives in VGPRs.  B_t/C_t for a 64-step chunk are
// staged cooperatively into LDS with GLOBAL_LOAD_ASYNC_TO_LDS_B128 (ASYNCcnt)
// and then read wave-uniform.  Per-thread dt/xc/z are batched 8 steps ahead
// of the serial recurrence.  Writes y into the dead xin half of xz (ld 4096).
// ---------------------------------------------------------------------------
#define SCHUNK 64

__global__ __launch_bounds__(256)
void scan_fused_kernel(const float* __restrict__ dt,
                       const float* __restrict__ xdbl,
                       const float* __restrict__ xc,
                       const float* __restrict__ xz,   // for z half
                       const float* __restrict__ A_log,
                       const float* __restrict__ Dp,
                       float* __restrict__ y)          // ld = 4096
{
    __shared__ float sBC[SCHUNK][32];   // [l][0:16)=B_t, [16:32)=C_t  (8 KB)

    const int tid = threadIdx.x;
    const int d = (blockIdx.x & 7) * 256 + tid;   // 8 blocks per batch entry
    const int b = blockIdx.x >> 3;

    float Arow[DS_SZ];
#pragma unroll
    for (int s = 0; s < DS_SZ; ++s) Arow[s] = -__expf(A_log[d * DS_SZ + s]);
    const float Dd = Dp[d];

    float h[DS_SZ];
#pragma unroll
    for (int s = 0; s < DS_SZ; ++s) h[s] = 0.0f;

    for (int lc = 0; lc < L_SZ; lc += SCHUNK) {
        // ---- async-stage the B/C chunk: 64 rows x 32 floats = 512 float4 --
#pragma unroll
        for (int i = 0; i < 2; ++i) {
            int e   = tid + i * 256;        // 0..511
            int row = e >> 3;               // 8 float4 per row
            int q   = (e & 7) << 2;
            const float* src =
                &xdbl[(size_t)(b * L_SZ + lc + row) * XP_SZ + DTR_SZ + q];
            unsigned lds_addr = (unsigned)(uintptr_t)&sBC[row][q];
            asm volatile("global_load_async_to_lds_b128 %0, %1, off"
                         :: "v"(lds_addr), "v"(src) : "memory");
        }
        asm volatile("s_wait_asynccnt 0x0" ::: "memory");
        __syncthreads();

        for (int j0 = 0; j0 < SCHUNK; j0 += 8) {
            // batch the per-thread scalars so loads overlap the serial chain
            float dt8[8], x8[8], z8[8];
#pragma unroll
            for (int u = 0; u < 8; ++u) {
                size_t rb = (size_t)(b * L_SZ + lc + j0 + u);
                dt8[u] = dt[rb * DI_SZ + d];
                x8[u]  = xc[rb * DI_SZ + d];
                z8[u]  = xz[rb * (2 * DI_SZ) + DI_SZ + d];
            }
#pragma unroll
            for (int u = 0; u < 8; ++u) {
                const int j = j0 + u;
                float Bv[DS_SZ], Cv[DS_SZ];
#pragma unroll
                for (int q = 0; q < 4; ++q) {
                    *reinterpret_cast<float4*>(&Bv[q * 4]) =
                        *reinterpret_cast<const float4*>(&sBC[j][q * 4]);
                    *reinterpret_cast<float4*>(&Cv[q * 4]) =
                        *reinterpret_cast<const float4*>(&sBC[j][16 + q * 4]);
                }
                const float dtv = dt8[u];
                const float xv  = x8[u];
                float ysum = 0.0f;
#pragma unroll
                for (int s = 0; s < DS_SZ; ++s) {
                    float dA = __expf(dtv * Arow[s]);
                    h[s] = dA * h[s] + dtv * Bv[s] * xv;
                    ysum += h[s] * Cv[s];
                }
                const float zv = z8[u];
                const float gate = zv / (1.0f + __expf(-zv));   // silu(z)
                const size_t rb = (size_t)(b * L_SZ + lc + j);
                y[rb * (2 * DI_SZ) + d] = (ysum + xv * Dd) * gate;
            }
        }
        __syncthreads();   // protect sBC before next chunk's async writes
    }
}

// ---------------------------------------------------------------------------
// Host launcher
// ---------------------------------------------------------------------------
extern "C" void kernel_launch(void* const* d_in, const int* in_sizes, int n_in,
                              void* d_out, int out_size, void* d_ws,
                              size_t ws_size, hipStream_t stream)
{
    const float* x      = (const float*)d_in[0];
    const float* W_in   = (const float*)d_in[1];
    const float* conv_w = (const float*)d_in[2];
    const float* conv_b = (const float*)d_in[3];
    const float* W_xprj = (const float*)d_in[4];
    const float* W_dt   = (const float*)d_in[5];
    const float* b_dt   = (const float*)d_in[6];
    const float* A_log  = (const float*)d_in[7];
    const float* Dp     = (const float*)d_in[8];
    const float* W_out  = (const float*)d_in[9];
    float* out = (float*)d_out;

    float* ws      = (float*)d_ws;
    float* ws_xz   = ws;                                   // 16384 x 4096
    float* ws_xc   = ws_xz  + (size_t)BL_SZ * 2 * DI_SZ;   // 16384 x 2048
    float* ws_xdbl = ws_xc  + (size_t)BL_SZ * DI_SZ;       // 16384 x 96
    float* ws_dt   = ws_xdbl + (size_t)BL_SZ * XP_SZ;      // 16384 x 2048

    dim3 blk(256);

    // 1) xz = x @ W_in      (16384 x 4096, K=1024)
    gemm_bf16_wmma<<<dim3((2 * DI_SZ) / TN, BL_SZ / TM), blk, 0, stream>>>(
        x, W_in, ws_xz, nullptr,
        BL_SZ, 2 * DI_SZ, DM_SZ, DM_SZ, 2 * DI_SZ, 2 * DI_SZ, 0);

    // 2) xc = silu(causal_dwconv(xin) + b)
    conv_silu_kernel<<<(BL_SZ * DI_SZ) / 256, blk, 0, stream>>>(
        ws_xz, conv_w, conv_b, ws_xc);

    // 3) x_dbl = xc @ W_xproj   (16384 x 96, K=2048)
    gemm_bf16_wmma<<<dim3(1, BL_SZ / TM), blk, 0, stream>>>(
        ws_xc, W_xprj, ws_xdbl, nullptr,
        BL_SZ, XP_SZ, DI_SZ, DI_SZ, XP_SZ, XP_SZ, 0);

    // 4) dt = softplus(x_dbl[:, :64] @ W_dt + b_dt)   (16384 x 2048, K=64)
    gemm_bf16_wmma<<<dim3(DI_SZ / TN, BL_SZ / TM), blk, 0, stream>>>(
        ws_xdbl, W_dt, ws_dt, b_dt,
        BL_SZ, DI_SZ, DTR_SZ, XP_SZ, DI_SZ, DI_SZ, 1);

    // 5) selective scan + D skip + silu(z) gate -> y (into xin half of xz)
    scan_fused_kernel<<<B_SZ * (DI_SZ / 256), blk, 0, stream>>>(
        ws_dt, ws_xdbl, ws_xc, ws_xz, A_log, Dp, ws_xz);

    // 6) out = y @ W_out     (16384 x 1024, K=2048, A ld=4096)
    gemm_bf16_wmma<<<dim3(DM_SZ / TN, BL_SZ / TM), blk, 0, stream>>>(
        ws_xz, W_out, out, nullptr,
        BL_SZ, DM_SZ, DI_SZ, 2 * DI_SZ, DM_SZ, DM_SZ, 0);
}
